// VectorQuantizeReduceSimple_27633819583045
// MI455X (gfx1250) — matrix operand order repaired and checked by reading, not compiled, not run on hardware
//
#include <hip/hip_runtime.h>
#include <hip/hip_bf16.h>

// ---------------------------------------------------------------------------
// VQ (reduce) on MI455X / gfx1250, wave32, bf16 WMMA 16x16x32 (K == e_red == 32)
// Pipeline:
//   k_norm_codes : cnorm (f32 + bf16) from code
//   k_cvt_w      : W_pre / W_post -> bf16
//   k_preproj    : z_reduce = z @ W_pre^T + b_pre (bf16 WMMA, f32 acc),
//                  zf = l2norm(z_reduce) stored bf16 (LDS transpose for stores)
//   k_vq         : scores = zf . cnorm^T via WMMA; code table double-buffered
//                  through LDS with global_load_async_to_lds_b128 (ASYNCcnt);
//                  running argmax at 3 VALU/score; idx -> d_out, loss partials
//   k_post       : z_q = (zr + (q - zr)) @ W_post^T + b_post (bf16 WMMA)
//   k_loss       : deterministic partial-sum reduction -> loss scalar
// ---------------------------------------------------------------------------

typedef __attribute__((ext_vector_type(16))) __bf16 v16bf;
typedef __attribute__((ext_vector_type(8)))  __bf16 v8bf;
typedef __attribute__((ext_vector_type(8)))  float  v8f;
typedef __attribute__((ext_vector_type(4)))  float  f32x4;

#define N_TOK   32768
#define E_DIM   1024
#define E_RED   32
#define N_CODE  8192
#define TPW     4          // token tiles per wave in k_vq
#define CHUNK_TILES 32     // code tiles staged per LDS chunk (512 rows, 32 KB)
#define N_CHUNK (N_CODE / (CHUNK_TILES * 16))   // 16 chunks

static __device__ __forceinline__ v8f wmma_bf16(v16bf a, v16bf b, v8f c) {
  // D = A(16x32 bf16) * B(32x16 bf16) + C(16x16 f32)
  return __builtin_amdgcn_wmma_f32_16x16x32_bf16(false, a, false, b,
                                                 (short)0, c, false, false);
}

// Async global->LDS copy, 16 bytes per lane (ASYNCcnt-tracked DMA path).
static __device__ __forceinline__ void async_copy16(unsigned ldsOff,
                                                    const void* gbase,
                                                    unsigned goff) {
  asm volatile("global_load_async_to_lds_b128 %0, %1, %2"
               :: "v"(ldsOff), "v"(goff), "s"(gbase)
               : "memory");
}
static __device__ __forceinline__ void wait_async0() {
  asm volatile("s_wait_asynccnt 0x0" ::: "memory");
}

// ---------------------------------------------------------------- K0a: codes
__global__ __launch_bounds__(256) void k_norm_codes(const float* __restrict__ code,
                                                    float* __restrict__ cf,
                                                    __bf16* __restrict__ cb) {
  int c = blockIdx.x * 256 + threadIdx.x;
  if (c >= N_CODE) return;
  const float* row = code + (size_t)c * E_RED;
  float v[E_RED];
#pragma unroll
  for (int i = 0; i < 8; ++i) ((f32x4*)v)[i] = ((const f32x4*)row)[i];
  float s = 0.f;
#pragma unroll
  for (int i = 0; i < E_RED; ++i) s += v[i] * v[i];
  float scale = 1.0f / fmaxf(sqrtf(s), 1e-12f);
  float o[E_RED];
#pragma unroll
  for (int i = 0; i < E_RED; ++i) o[i] = v[i] * scale;
  float* ofr = cf + (size_t)c * E_RED;
#pragma unroll
  for (int i = 0; i < 8; ++i) ((f32x4*)ofr)[i] = ((f32x4*)o)[i];
  __bf16 ob[E_RED];
#pragma unroll
  for (int i = 0; i < E_RED; ++i) ob[i] = (__bf16)o[i];
  __bf16* obr = cb + (size_t)c * E_RED;
#pragma unroll
  for (int i = 0; i < 4; ++i) ((f32x4*)obr)[i] = ((f32x4*)ob)[i];
}

// ------------------------------------------------------------- K0b: weights
__global__ __launch_bounds__(256) void k_cvt_w(const float* __restrict__ wpre,
                                               const float* __restrict__ wpost,
                                               __bf16* __restrict__ wpreb,
                                               __bf16* __restrict__ wpostb) {
  int i = blockIdx.x * 256 + threadIdx.x;     // 65536 total
  if (i < E_RED * E_DIM) wpreb[i]  = (__bf16)wpre[i];
  else                   wpostb[i - E_RED * E_DIM] = (__bf16)wpost[i - E_RED * E_DIM];
}

// ---------------------------------------------------- K1: pre-proj + l2norm
__global__ __launch_bounds__(256) void k_preproj(const float* __restrict__ z,
                                                 const __bf16* __restrict__ wpreb,
                                                 const float* __restrict__ bpre,
                                                 float* __restrict__ zred,
                                                 __bf16* __restrict__ zfb) {
  __shared__ float tile[8][16][32];           // 16 KB, per-wave transpose buffer
  const int w    = threadIdx.x >> 5;          // wave in block
  const int wave = (blockIdx.x << 3) + w;     // 0..2047 token tiles
  const int lane = threadIdx.x & 31;
  const int li   = lane & 15, hi = lane >> 4;

  const float* arow = z + (size_t)(wave * 16 + li) * E_DIM;   // token row (A, M=li)
  v8f c0 = {}, c1 = {};
  for (int kb = 0; kb < E_DIM; kb += 32) {
    // A fragment: f32 -> bf16, 16 contiguous K at kb + 16*hi
    float t1[8], t2[8];
    const float* r1 = arow + kb + hi * 16;
    const float* r2 = arow + kb + hi * 16 + 8;
    ((f32x4*)t1)[0] = ((const f32x4*)r1)[0]; ((f32x4*)t1)[1] = ((const f32x4*)r1)[1];
    ((f32x4*)t2)[0] = ((const f32x4*)r2)[0]; ((f32x4*)t2)[1] = ((const f32x4*)r2)[1];
    v16bf a;
#pragma unroll
    for (int i = 0; i < 8; ++i) { a[i] = (__bf16)t1[i]; a[8 + i] = (__bf16)t2[i]; }
    // B fragments: W_pre^T, lane n = li (+16), 16 contiguous K at kb + 16*hi
    v16bf b0 = *(const v16bf*)(wpreb + (size_t)li        * E_DIM + kb + hi * 16);
    v16bf b1 = *(const v16bf*)(wpreb + (size_t)(li + 16) * E_DIM + kb + hi * 16);
    c0 = wmma_bf16(a, b0, c0);
    c1 = wmma_bf16(a, b1, c1);
  }
  // scatter C (M=r+8*hi, N=li) into LDS [token][n]
#pragma unroll
  for (int r = 0; r < 8; ++r) {
    tile[w][r + hi * 8][li]      = c0[r];
    tile[w][r + hi * 8][li + 16] = c1[r];
  }
  __syncthreads();
  // lane handles (token=li, half=hi): bias, l2norm, coalesced stores
  float f[16];
#pragma unroll
  for (int i = 0; i < 16; ++i) f[i] = tile[w][li][hi * 16 + i] + bpre[hi * 16 + i];
  float s = 0.f;
#pragma unroll
  for (int i = 0; i < 16; ++i) s += f[i] * f[i];
  s += __shfl_xor(s, 16);
  float scale = 1.0f / fmaxf(sqrtf(s), 1e-12f);
  float* zr = zred + (size_t)(wave * 16 + li) * E_RED + hi * 16;
#pragma unroll
  for (int i = 0; i < 4; ++i) ((f32x4*)zr)[i] = ((f32x4*)f)[i];
  __bf16 zb[16];
#pragma unroll
  for (int i = 0; i < 16; ++i) zb[i] = (__bf16)(f[i] * scale);
  *(v16bf*)(zfb + (size_t)(wave * 16 + li) * E_RED + hi * 16) = *(v16bf*)zb;
}

// -------------------------------------------------- K2: argmax + loss parts
__global__ __launch_bounds__(256) void k_vq(const __bf16* __restrict__ zfb,
                                            const __bf16* __restrict__ cb,
                                            const float* __restrict__ cf,
                                            const float* __restrict__ zred,
                                            int* __restrict__ idx_out,
                                            float* __restrict__ partial) {
  __shared__ __bf16 cbuf[2][CHUNK_TILES * 16 * E_RED];     // 2 x 32 KB
  const int t    = threadIdx.x;
  const int wave = blockIdx.x * 8 + (t >> 5);              // 0..511
  const int lane = t & 31;
  const int li = lane & 15, hi = lane >> 4;
  const int tok0 = wave * (16 * TPW);
  const unsigned CHUNK_BYTES = CHUNK_TILES * 16 * E_RED * 2;   // 32768

  v16bf bfr[TPW];                                          // zf: B fragments (N=tokens)
#pragma unroll
  for (int j = 0; j < TPW; ++j)
    bfr[j] = *(const v16bf*)(zfb + (size_t)(tok0 + j * 16 + li) * E_RED + hi * 16);

  float best[TPW]; int bp[TPW];
#pragma unroll
  for (int j = 0; j < TPW; ++j) { best[j] = -3.0e38f; bp[j] = 0; }

  // prologue: stage chunk 0 into buffer 0 (16 B per lane per copy)
#pragma unroll
  for (int it = 0; it < 8; ++it) {
    unsigned off = (unsigned)(it * 256 + t) * 16u;
    async_copy16((unsigned)(uintptr_t)((char*)&cbuf[0][0] + off), cb, off);
  }
  wait_async0();
  __syncthreads();

  for (int chunk = 0; chunk < N_CHUNK; ++chunk) {
    const int buf = chunk & 1;
    // kick off the next chunk's DMA into the other buffer
    if (chunk + 1 < N_CHUNK) {
      unsigned gbase = (unsigned)(chunk + 1) * CHUNK_BYTES;
#pragma unroll
      for (int it = 0; it < 8; ++it) {
        unsigned off = (unsigned)(it * 256 + t) * 16u;
        async_copy16((unsigned)(uintptr_t)((char*)&cbuf[buf ^ 1][0] + off),
                     cb, gbase + off);
      }
    }
    const __bf16* cl = &cbuf[buf][0];
    const int ctBase = chunk * CHUNK_TILES;
    for (int ctl = 0; ctl < CHUNK_TILES; ++ctl) {
      const __bf16* cr = cl + (ctl * 16 + li) * E_RED;     // A: code M=li (LDS)
      v8bf a1 = *(const v8bf*)(cr + hi * 16);
      v8bf a2 = *(const v8bf*)(cr + hi * 16 + 8);
      v16bf a;
#pragma unroll
      for (int i = 0; i < 8; ++i) { a[i] = a1[i]; a[8 + i] = a2[i]; }
      const int sbase = (ctBase + ctl) * 16;               // uniform -> SGPR
#pragma unroll
      for (int j = 0; j < TPW; ++j) {
        v8f zc = {};
        v8f d = wmma_bf16(a, bfr[j], zc);                  // D[M=code, N=token]
#pragma unroll
        for (int r = 0; r < 8; ++r) {                      // 3 VALU per score
          float sc = d[r];
          bool gt = sc > best[j];
          best[j] = gt ? sc : best[j];
          bp[j]   = gt ? (sbase + r) : bp[j];              // scalar operand
        }
      }
    }
    if (chunk + 1 < N_CHUNK) {
      wait_async0();            // own copies complete ...
      __syncthreads();          // ... and everyone's; buffer handoff
    }
  }

  float lsum = 0.0f;
#pragma unroll
  for (int j = 0; j < TPW; ++j) {
    int   myI = bp[j] + hi * 8;                            // recover lane part
    float myS = best[j];
    float oS = __shfl_xor(myS, 16);
    int   oI = __shfl_xor(myI, 16);
    int   fI = (oS > myS || (oS == myS && oI < myI)) ? oI : myI;
    int token = tok0 + j * 16 + li;
    if (hi == 0) idx_out[token] = fI;
    // loss contribution: lane covers half 'hi' of the 32 components
    const float* q  = cf   + (size_t)fI    * E_RED + hi * 16;
    const float* zr = zred + (size_t)token * E_RED + hi * 16;
#pragma unroll
    for (int i = 0; i < 16; ++i) { float dd = q[i] - zr[i]; lsum += dd * dd; }
  }
#pragma unroll
  for (int m = 16; m >= 1; m >>= 1) lsum += __shfl_xor(lsum, m);
  if (lane == 0) partial[wave] = lsum;
}

// --------------------------------------------------------- K3: post-project
__global__ __launch_bounds__(256) void k_post(const float* __restrict__ cf,
                                              const float* __restrict__ zred,
                                              const int* __restrict__ idx,
                                              const __bf16* __restrict__ wpostb,
                                              const float* __restrict__ bpost,
                                              float* __restrict__ zq) {
  const int wave = blockIdx.x * 8 + (threadIdx.x >> 5);    // 0..2047 token tiles
  const int lane = threadIdx.x & 31;
  const int li = lane & 15, hi = lane >> 4;
  const int token = wave * 16 + li;
  const int id = idx[token];
  const float* q  = cf   + (size_t)id    * E_RED;
  const float* zr = zred + (size_t)token * E_RED;
  // straight-through forward: a = zr + (q - zr), 16 contiguous K at 16*hi
  float q1[8], q2[8], z1[8], z2[8];
  ((f32x4*)q1)[0] = ((const f32x4*)(q  + hi * 16))[0];  ((f32x4*)q1)[1] = ((const f32x4*)(q  + hi * 16))[1];
  ((f32x4*)q2)[0] = ((const f32x4*)(q  + hi * 16 + 8))[0]; ((f32x4*)q2)[1] = ((const f32x4*)(q  + hi * 16 + 8))[1];
  ((f32x4*)z1)[0] = ((const f32x4*)(zr + hi * 16))[0];  ((f32x4*)z1)[1] = ((const f32x4*)(zr + hi * 16))[1];
  ((f32x4*)z2)[0] = ((const f32x4*)(zr + hi * 16 + 8))[0]; ((f32x4*)z2)[1] = ((const f32x4*)(zr + hi * 16 + 8))[1];
  v16bf a;
#pragma unroll
  for (int i = 0; i < 8; ++i) {
    a[i]     = (__bf16)(z1[i] + (q1[i] - z1[i]));
    a[8 + i] = (__bf16)(z2[i] + (q2[i] - z2[i]));
  }
  for (int nt = 0; nt < E_DIM / 16; ++nt) {
    int n = nt * 16 + li;
    v16bf b = *(const v16bf*)(wpostb + (size_t)n * E_RED + hi * 16);  // W_post^T
    v8f zc = {};
    v8f d = wmma_bf16(a, b, zc);                           // D[M=token, N=dim]
    float bb = bpost[n];
    float* out = zq + (size_t)(wave * 16 + hi * 8) * E_DIM + n;
#pragma unroll
    for (int r = 0; r < 8; ++r)
      out[(size_t)r * E_DIM] = d[r] + bb;
  }
}

// ------------------------------------------------------------ K4: loss sum
__global__ __launch_bounds__(256) void k_loss(const float* __restrict__ partial,
                                              float* __restrict__ loss_out) {
  __shared__ float sm[256];
  float s = partial[threadIdx.x] + partial[threadIdx.x + 256];   // 512 partials
  sm[threadIdx.x] = s;
  __syncthreads();
  for (int st = 128; st > 0; st >>= 1) {
    if (threadIdx.x < (unsigned)st) sm[threadIdx.x] += sm[threadIdx.x + st];
    __syncthreads();
  }
  if (threadIdx.x == 0)
    loss_out[0] = sm[0] * (1.25f / ((float)N_TOK * (float)E_RED));
}

// ---------------------------------------------------------------- launcher
extern "C" void kernel_launch(void* const* d_in, const int* in_sizes, int n_in,
                              void* d_out, int out_size, void* d_ws, size_t ws_size,
                              hipStream_t stream) {
  (void)in_sizes; (void)n_in; (void)out_size; (void)ws_size;
  const float* z     = (const float*)d_in[0];
  const float* code  = (const float*)d_in[1];
  const float* wpre  = (const float*)d_in[2];
  const float* bpre  = (const float*)d_in[3];
  const float* wpost = (const float*)d_in[4];
  const float* bpost = (const float*)d_in[5];

  char* ws = (char*)d_ws;                                   // ~8 MB + 2 KB used
  float*  zred   = (float*) (ws);                           // 4 MB
  __bf16* zfb    = (__bf16*)(ws + (4u  << 20));             // 2 MB
  float*  cf     = (float*) (ws + (6u  << 20));             // 1 MB
  __bf16* cb     = (__bf16*)(ws + (7u  << 20));             // 512 KB
  __bf16* wpreb  = (__bf16*)(ws + (7u  << 20) + (512u << 10)); // 64 KB
  __bf16* wpostb = wpreb + E_RED * E_DIM;                   // 64 KB
  float*  partial= (float*) (ws + (8u  << 20));             // 2 KB

  float* zq   = (float*)d_out;                              // [8,4096,1024]
  float* loss = zq + (size_t)N_TOK * E_DIM;                 // scalar
  int*   idx  = (int*)(loss + 1);                           // [32768] int32

  k_norm_codes<<<N_CODE / 256, 256, 0, stream>>>(code, cf, cb);
  k_cvt_w     <<<(2 * E_RED * E_DIM) / 256, 256, 0, stream>>>(wpre, wpost, wpreb, wpostb);
  k_preproj   <<<N_TOK / (16 * 8), 256, 0, stream>>>(z, wpreb, bpre, zred, zfb);
  k_vq        <<<N_TOK / (16 * TPW * 8), 256, 0, stream>>>(zfb, cb, cf, zred, idx, partial);
  k_post      <<<N_TOK / (16 * 8), 256, 0, stream>>>(cf, zred, idx, wpostb, bpost, zq);
  k_loss      <<<1, 256, 0, stream>>>(partial, loss);
}
